// AgentEmbedding_44418551775449
// MI455X (gfx1250) — matrix operand order repaired and checked by reading, not compiled, not run on hardware
//
#include <hip/hip_runtime.h>

#define BSZ   8
#define SLEN  512
#define DV    64
#define DH    128
#define DHID  256
#define NSEQ  (BSZ*SLEN)        // 4096
#define KTOT  (DV+DH)           // 192
#define G4    (4*DH)            // 512
#define LN_EPS 1e-5f

typedef __bf16 bf16;
typedef __attribute__((ext_vector_type(16))) __bf16 v16bf;
typedef __attribute__((ext_vector_type(8)))  __bf16 v8bf;
typedef __attribute__((ext_vector_type(8)))  float  v8f;

// ---- workspace layout (bytes) ----
#define WS_WFRAG  0                         // 32*6*32*16 bf16 = 196608 B
#define WS_CFRAG  196608                    // 16*8*32*16 bf16 = 131072 B
#define WS_BIAS   (196608 + 131072)         // 512 f32 = 2048 B
#define WS_BW     (196608 + 131072 + 2048)  // 4096*128 f32 = 2 MB

static __device__ inline v8f vzero8() {
  v8f z;
#pragma unroll
  for (int i = 0; i < 8; ++i) z[i] = 0.f;
  return z;
}

static __device__ inline float sigm(float x) { return 1.f / (1.f + __expf(-x)); }
static __device__ inline float tanh_f(float x) {
  x = fminf(fmaxf(x, -15.f), 15.f);
  float e = __expf(2.f * x);
  return (e - 1.f) / (e + 1.f);
}

// Gather one 16x32 bf16 A-fragment from a row-major LDS tile, per the
// documented 16-bit A VGPR layout: lanes 0-15 hold K = [0..7, 16..23] of
// row (lane), lanes 16-31 hold K = [8..15, 24..31] of row (lane-16).
static __device__ inline v16bf load_a_frag(const bf16* Abase, int lane,
                                           int rowStride, int kb) {
  const int row = lane & 15;
  const v8bf lo = *(const v8bf*)(Abase + row * rowStride + kb);
  const v8bf hi = *(const v8bf*)(Abase + row * rowStride + kb + 16);
  v16bf a;
#pragma unroll
  for (int e = 0; e < 8; ++e) { a[e] = lo[e]; a[e + 8] = hi[e]; }
  return a;
}

// -------------------------------------------------------------------------
// Prep: bf16-convert + swizzle weights into WMMA B-fragment order
// (dense 16-bit B 32x16: lane n<16 holds col n, K=kc*32+[0..7,16..23]; lane>=16 holds
// col n-16, K=kc*32+[8..15,24..31]), and fold the two bias vectors.
// -------------------------------------------------------------------------
__global__ __launch_bounds__(256) void prep_kernel(
    const float* __restrict__ W_ih, const float* __restrict__ W_hh,
    const float* __restrict__ conv_w, const float* __restrict__ b_ih,
    const float* __restrict__ b_hh, bf16* __restrict__ wfrag,
    bf16* __restrict__ cfrag, float* __restrict__ bias) {
  const int tid = blockIdx.x * 256 + threadIdx.x;
  if (tid < 32 * 6 * 32 * 16) {               // LSTM combined weights: K=192, N=512
    const int e = tid & 15, lane = (tid >> 4) & 31;
    const int f = tid >> 9;                   // ct*6 + kc
    const int kc = f % 6, ct = f / 6;
    const int n = ct * 16 + (lane & 15);
    const int k = kc * 32 + ((lane >= 16) ? 8 : 0) + ((e < 8) ? e : e + 8);
    const float v = (k < DV) ? W_ih[n * DV + k] : W_hh[n * DH + (k - DV)];
    wfrag[tid] = (bf16)v;
  } else if (tid < 32 * 6 * 32 * 16 + 16 * 8 * 32 * 16) {  // conv: K=256, N=256
    const int q = tid - 32 * 6 * 32 * 16;
    const int e = q & 15, lane = (q >> 4) & 31;
    const int f = q >> 9;                     // ct*8 + kc
    const int kc = f & 7, ct = f >> 3;
    const int n = ct * 16 + (lane & 15);
    const int k = kc * 32 + ((lane >= 16) ? 8 : 0) + ((e < 8) ? e : e + 8);
    cfrag[q] = (bf16)conv_w[n * DHID + k];
  } else if (tid < 32 * 6 * 32 * 16 + 16 * 8 * 32 * 16 + G4) {
    const int q = tid - (32 * 6 * 32 * 16 + 16 * 8 * 32 * 16);
    bias[q] = b_ih[q] + b_hh[q];
  }
}

// -------------------------------------------------------------------------
// LSTM scan. One block = 16 sequences, 512 threads = 16 waves.
// Each wave owns 2 gate column-tiles; its 12 B-fragments stay in VGPRs for
// the entire 512-step scan. Cell state c stays in thread registers.
// -------------------------------------------------------------------------
__global__ __launch_bounds__(512) void lstm_scan_kernel(
    const float* __restrict__ xw,   // (4096, 512, 64)
    const int* __restrict__ xn,     // (4096)
    const bf16* __restrict__ wfrag, // [32][6][32][16]
    const float* __restrict__ bias, // (512)
    float* __restrict__ bw) {       // (4096, 128)
  __shared__ bf16  A_lds[16 * KTOT];   // [row][k]  : [x_t (64) | h (128)] bf16
  __shared__ float g_lds[16 * G4];     // [row][gate col] f32

  const int tid  = threadIdx.x;
  const int lane = tid & 31;
  const int wv   = tid >> 5;            // 0..15
  const int seq0 = blockIdx.x * 16;

  // B fragments -> registers (reused for all 512 steps)
  v16bf B[2][6];
#pragma unroll
  for (int i = 0; i < 2; ++i) {
    const int ct = wv * 2 + i;
#pragma unroll
    for (int kc = 0; kc < 6; ++kc)
      B[i][kc] = *(const v16bf*)(wfrag + (((size_t)(ct * 6 + kc) * 32 + lane) << 4));
  }

  // Elementwise ownership: wave wv owns row wv; lane owns 4 hidden units.
  const int r  = wv;
  const int jb = lane * 4;
  float c_reg[4] = {0.f, 0.f, 0.f, 0.f};
  float bI[4], bF[4], bG[4], bO[4];
#pragma unroll
  for (int u = 0; u < 4; ++u) {
    bI[u] = bias[jb + u];
    bF[u] = bias[DH + jb + u];
    bG[u] = bias[2 * DH + jb + u];
    bO[u] = bias[3 * DH + jb + u];
  }
  const int t_last = xn[seq0 + r] - 1;
  float* bw_row = bw + (size_t)(seq0 + r) * DH + jb;

  for (int i = tid; i < 16 * KTOT; i += 512) A_lds[i] = (bf16)0.0f;

  // x staging: thread loads 2 floats of the 16x64 tile each step
  const int xr = tid >> 5;
  const int xd = lane * 2;
  const float* xbase = xw + (size_t)(seq0 + xr) * SLEN * DV + xd;
  __syncthreads();

  for (int t = 0; t < SLEN; ++t) {
    // ---- stage x_t (f32 -> bf16) into A_lds[:, 0:64] ----
    const float* xp = xbase + (size_t)t * DV;
    const float2 xv = *(const float2*)xp;
    bf16* ax = &A_lds[xr * KTOT + xd];
    ax[0] = (bf16)xv.x;
    ax[1] = (bf16)xv.y;
    // Prefetch next step's x_t with high locality -> WGP-scope
    // global_prefetch_b8 (pull into all cache levels; consumed next iter).
    if (t + 1 < SLEN) __builtin_prefetch(xp + DV, 0, 3);
    __syncthreads();  // x_t + h(prev) visible; g_lds free to overwrite

    // ---- g = [x_t, h] @ Wcomb^T : 12 WMMAs per wave ----
    v8f acc0 = vzero8(), acc1 = vzero8();
#pragma unroll
    for (int kc = 0; kc < 6; ++kc) {
      const int kb = kc * 32 + ((lane >= 16) ? 8 : 0);
      const v16bf a = load_a_frag(A_lds, lane, KTOT, kb);
      acc0 = __builtin_amdgcn_wmma_f32_16x16x32_bf16(false, a, false, B[0][kc],
                                                     (short)0, acc0, false, false);
      acc1 = __builtin_amdgcn_wmma_f32_16x16x32_bf16(false, a, false, B[1][kc],
                                                     (short)0, acc1, false, false);
    }
    // scatter per C-layout: VGPR v -> row v + (lane>=16?8:0), col ct*16 + lane%16
    {
      const int m0 = (lane >= 16) ? 8 : 0;
      const int cl = lane & 15;
      float* g0 = &g_lds[m0 * G4 + (wv * 2 + 0) * 16 + cl];
      float* g1 = &g_lds[m0 * G4 + (wv * 2 + 1) * 16 + cl];
#pragma unroll
      for (int v = 0; v < 8; ++v) {
        g0[v * G4] = acc0[v];
        g1[v * G4] = acc1[v];
      }
    }
    __syncthreads();  // gates ready

    // ---- elementwise LSTM cell update (torch gate order i,f,g,o) ----
    const float* gr = &g_lds[r * G4];
    const float4 gi = *(const float4*)&gr[jb];
    const float4 gf = *(const float4*)&gr[DH + jb];
    const float4 gg = *(const float4*)&gr[2 * DH + jb];
    const float4 go = *(const float4*)&gr[3 * DH + jb];
    const float iv[4] = {gi.x, gi.y, gi.z, gi.w};
    const float fv[4] = {gf.x, gf.y, gf.z, gf.w};
    const float gv[4] = {gg.x, gg.y, gg.z, gg.w};
    const float ov[4] = {go.x, go.y, go.z, go.w};
    float hv[4];
#pragma unroll
    for (int u = 0; u < 4; ++u) {
      const float is = sigm(iv[u] + bI[u]);
      const float fs = sigm(fv[u] + bF[u]);
      const float os = sigm(ov[u] + bO[u]);
      const float gt = tanh_f(gv[u] + bG[u]);
      const float c  = fs * c_reg[u] + is * gt;
      c_reg[u] = c;
      const float h = os * tanh_f(c);
      hv[u] = h;
      A_lds[r * KTOT + DV + jb + u] = (bf16)h;  // feed next step's A
    }
    if (t == t_last) {
      float4 o;
      o.x = hv[0]; o.y = hv[1]; o.z = hv[2]; o.w = hv[3];
      *(float4*)bw_row = o;
    }
    // next iteration's first barrier orders h/g_lds reuse
  }
}

// -------------------------------------------------------------------------
// Tail: h = [er | egw + bw] -> y = h @ conv_w^T + conv_b -> LayerNorm.
// One block = 16 rows, 256 threads = 8 waves, each wave 2 col tiles.
// -------------------------------------------------------------------------
__global__ __launch_bounds__(256) void conv_ln_kernel(
    const float* __restrict__ er, const float* __restrict__ egw,
    const float* __restrict__ bwv, const bf16* __restrict__ cfrag,
    const float* __restrict__ conv_b, const float* __restrict__ ln_g,
    const float* __restrict__ ln_b, float* __restrict__ out) {
  __shared__ bf16  A_lds[16 * DHID];
  __shared__ float y_lds[16 * DHID];

  const int tid  = threadIdx.x;
  const int lane = tid & 31;
  const int wv   = tid >> 5;
  const int row0 = blockIdx.x * 16;

  // stage A (bf16): 16 contiguous elements per thread
#pragma unroll
  for (int i = 0; i < 16; ++i) {
    const int n = tid * 16 + i;
    const int rr = n >> 8, k = n & 255;
    const size_t g = (size_t)(row0 + rr);
    const float v = (k < DH) ? er[g * DH + k]
                             : (egw[g * DH + (k - DH)] + bwv[g * DH + (k - DH)]);
    A_lds[n] = (bf16)v;
  }
  __syncthreads();

  v8f acc0 = vzero8(), acc1 = vzero8();
#pragma unroll
  for (int kc = 0; kc < 8; ++kc) {
    const int kb = kc * 32 + ((lane >= 16) ? 8 : 0);
    const v16bf a = load_a_frag(A_lds, lane, DHID, kb);
    const v16bf b0 = *(const v16bf*)(cfrag + (((size_t)((wv * 2 + 0) * 8 + kc) * 32 + lane) << 4));
    const v16bf b1 = *(const v16bf*)(cfrag + (((size_t)((wv * 2 + 1) * 8 + kc) * 32 + lane) << 4));
    acc0 = __builtin_amdgcn_wmma_f32_16x16x32_bf16(false, a, false, b0, (short)0, acc0, false, false);
    acc1 = __builtin_amdgcn_wmma_f32_16x16x32_bf16(false, a, false, b1, (short)0, acc1, false, false);
  }
  {
    const int m0 = (lane >= 16) ? 8 : 0;
    const int cl = lane & 15;
    const int c0 = (wv * 2 + 0) * 16 + cl, c1 = (wv * 2 + 1) * 16 + cl;
    const float cb0 = conv_b[c0], cb1 = conv_b[c1];
#pragma unroll
    for (int v = 0; v < 8; ++v) {
      y_lds[(m0 + v) * DHID + c0] = acc0[v] + cb0;
      y_lds[(m0 + v) * DHID + c1] = acc1[v] + cb1;
    }
  }
  __syncthreads();

  // LayerNorm: 16 lanes per row (lane groups of 16 align with wave halves)
  const int r  = tid >> 4;
  const int cb = (tid & 15) * 16;
  float s = 0.f, s2 = 0.f, vals[16];
#pragma unroll
  for (int u = 0; u < 16; ++u) {
    const float v = y_lds[r * DHID + cb + u];
    vals[u] = v;
    s += v;
    s2 += v * v;
  }
#pragma unroll
  for (int m = 1; m < 16; m <<= 1) {
    s  += __shfl_xor(s, m, 16);
    s2 += __shfl_xor(s2, m, 16);
  }
  const float mu = s * (1.f / DHID);
  const float var = s2 * (1.f / DHID) - mu * mu;
  const float rs = rsqrtf(var + LN_EPS);
  float* orow = out + (size_t)(row0 + r) * DHID + cb;
#pragma unroll
  for (int u = 0; u < 16; ++u)
    orow[u] = (vals[u] - mu) * rs * ln_g[cb + u] + ln_b[cb + u];
}

extern "C" void kernel_launch(void* const* d_in, const int* in_sizes, int n_in,
                              void* d_out, int out_size, void* d_ws, size_t ws_size,
                              hipStream_t stream) {
  const float* xw     = (const float*)d_in[0];
  const int*   xn     = (const int*)d_in[1];
  const float* er     = (const float*)d_in[2];
  const float* egw    = (const float*)d_in[3];
  const float* W_ih   = (const float*)d_in[4];
  const float* W_hh   = (const float*)d_in[5];
  const float* b_ih   = (const float*)d_in[6];
  const float* b_hh   = (const float*)d_in[7];
  const float* conv_w = (const float*)d_in[8];
  const float* conv_b = (const float*)d_in[9];
  const float* ln_g   = (const float*)d_in[10];
  const float* ln_b   = (const float*)d_in[11];
  float* out = (float*)d_out;

  char* ws = (char*)d_ws;
  bf16*  wfrag = (bf16*)(ws + WS_WFRAG);
  bf16*  cfrag = (bf16*)(ws + WS_CFRAG);
  float* bias  = (float*)(ws + WS_BIAS);
  float* bw    = (float*)(ws + WS_BW);

  const int prep_n = 32 * 6 * 32 * 16 + 16 * 8 * 32 * 16 + G4;
  prep_kernel<<<(prep_n + 255) / 256, 256, 0, stream>>>(
      W_ih, W_hh, conv_w, b_ih, b_hh, wfrag, cfrag, bias);

  lstm_scan_kernel<<<NSEQ / 16, 512, 0, stream>>>(xw, xn, wfrag, bias, bw);

  conv_ln_kernel<<<NSEQ / 16, 256, 0, stream>>>(
      er, egw, bw, cfrag, conv_b, ln_g, ln_b, out);
}
